// GNNModule_88931592831411
// MI455X (gfx1250) — matrix operand order repaired and checked by reading, not compiled, not run on hardware
//
#include <hip/hip_runtime.h>

#define FEAT 128
#define BN_EPS 1e-5f

typedef __attribute__((ext_vector_type(16))) _Float16 v16h;
typedef __attribute__((ext_vector_type(8)))  float    v8f;

// ---------- degree / norm ----------
__global__ void init_deg_kernel(float* __restrict__ deg, int N) {
  int i = blockIdx.x * blockDim.x + threadIdx.x;
  if (i < N) deg[i] = 1.0f;  // self-loop contributes 1
}

__global__ void accum_deg_kernel(const int* __restrict__ dst, float* __restrict__ deg, int E) {
  int e = blockIdx.x * blockDim.x + threadIdx.x;
  if (e < E) atomicAdd(&deg[dst[e]], 1.0f);
}

__global__ void deg_to_dis_kernel(float* __restrict__ deg, int N) {
  int i = blockIdx.x * blockDim.x + threadIdx.x;
  if (i < N) deg[i] = rsqrtf(fmaxf(deg[i], 1e-12f));
}

// ---------- dense projection: out[N,128] = A[N,128] @ W[128,128] ----------
// Split-f16 emulated fp32 GEMM on v_wmma_f32_16x16x32_f16:
//   A = Ahi + Alo, B = Bhi + Blo (f16 splits), D += Ahi*Bhi + Ahi*Blo + Alo*Bhi (f32 acc).
// One block = 256 threads = 8 waves; each wave computes a 16x128 output strip.
// W pre-split into f16 hi/lo LDS buffers laid out in B-fragment order so each
// lane's v16h fragment is a single contiguous 32B LDS load.
__global__ __launch_bounds__(256)
void gemm_wmma_kernel(const float* __restrict__ A, const float* __restrict__ W,
                      float* __restrict__ out, int N) {
  __shared__ __attribute__((aligned(32))) _Float16 Whi[FEAT * FEAT];  // 32 KB
  __shared__ __attribute__((aligned(32))) _Float16 Wlo[FEAT * FEAT];  // 32 KB

  // Fragment-order fill: index i = ((ks*8 + nt)*32 + lane)*16 + e
  //   K = ks*32 + (e>>3)*16 + (lane>>4)*8 + (e&7) ; n = nt*16 + (lane&15)
  for (int i = threadIdx.x; i < FEAT * FEAT; i += 256) {
    int e    = i & 15;
    int ln   = (i >> 4) & 31;
    int nt   = (i >> 9) & 7;
    int ks   = i >> 12;
    int K    = ks * 32 + (e >> 3) * 16 + (ln >> 4) * 8 + (e & 7);
    int n    = nt * 16 + (ln & 15);
    float w  = W[K * FEAT + n];
    _Float16 hi = (_Float16)w;
    Whi[i] = hi;
    Wlo[i] = (_Float16)(w - (float)hi);
  }
  __syncthreads();

  const int lane = threadIdx.x & 31;
  const int h    = lane >> 4;   // half-wave select
  const int l16  = lane & 15;
  const int wv   = threadIdx.x >> 5;

  const long row0 = (long)blockIdx.x * 128 + (long)wv * 16;
  long r = row0 + l16;
  if (r >= N) r = N - 1;              // clamp loads; stores guarded (EXEC stays all-1s)
  const float* arow = A + r * FEAT;

  v8f acc[8] = {};
  for (int ks = 0; ks < 4; ks++) {
    const int kk = ks * 32;
    // A-frag per ISA 16x32 f16 layout: e<8 -> K=kk+h*8+e ; e>=8 -> K=kk+16+h*8+(e-8)
    v16h ahi, alo;
#pragma unroll
    for (int e = 0; e < 8; e++) {
      float v0 = arow[kk + h * 8 + e];
      _Float16 h0 = (_Float16)v0;
      ahi[e] = h0;
      alo[e] = (_Float16)(v0 - (float)h0);
      float v1 = arow[kk + 16 + h * 8 + e];
      _Float16 h1 = (_Float16)v1;
      ahi[e + 8] = h1;
      alo[e + 8] = (_Float16)(v1 - (float)h1);
    }
#pragma unroll
    for (int nt = 0; nt < 8; nt++) {
      const int fo = ((ks * 8 + nt) * 32 + lane) * 16;
      v16h bhi = *(const v16h*)(Whi + fo);
      v16h blo = *(const v16h*)(Wlo + fo);
      acc[nt] = __builtin_amdgcn_wmma_f32_16x16x32_f16(
          false, ahi, false, bhi, (short)0, acc[nt], false, false);
      acc[nt] = __builtin_amdgcn_wmma_f32_16x16x32_f16(
          false, ahi, false, blo, (short)0, acc[nt], false, false);
      acc[nt] = __builtin_amdgcn_wmma_f32_16x16x32_f16(
          false, alo, false, bhi, (short)0, acc[nt], false, false);
    }
  }
  // D layout: VGPR j -> row (row0+j) for lanes 0-15, row (row0+j+8) for lanes 16-31
#pragma unroll
  for (int nt = 0; nt < 8; nt++) {
#pragma unroll
    for (int j = 0; j < 8; j++) {
      long rr = row0 + j + h * 8;
      if (rr < N) out[rr * FEAT + nt * 16 + l16] = acc[nt][j];
    }
  }
}

// ---------- agg init: bias + self-loop term (agg = b + m * dis^2), no atomics ----------
__global__ void init_agg_kernel(const float* __restrict__ m, const float* __restrict__ dis,
                                const float* __restrict__ bias, float* __restrict__ agg,
                                long total) {
  long idx = (long)blockIdx.x * blockDim.x + threadIdx.x;
  if (idx >= total) return;
  long n = idx >> 7;
  int  f = (int)(idx & (FEAT - 1));
  float d = dis[n];
  agg[idx] = bias[f] + m[idx] * d * d;
}

// ---------- edge scatter: agg[dst] += m[src] * dis[src]*dis[dst] ----------
// 256-thread block handles 2 edges; 128 threads (one per feature) per edge.
__global__ __launch_bounds__(256)
void scatter_edges_kernel(const float* __restrict__ m, const int* __restrict__ src,
                          const int* __restrict__ dst, const float* __restrict__ dis,
                          float* __restrict__ agg, int E) {
  int e = blockIdx.x * 2 + (threadIdx.x >> 7);
  if (e >= E) return;
  int f = threadIdx.x & (FEAT - 1);
  int s = src[e];
  int d = dst[e];
  float w = dis[s] * dis[d];
  float v = m[(long)s * FEAT + f] * w;
  atomicAdd(&agg[(long)d * FEAT + f], v);
}

// ---------- batchnorm ----------
__global__ void zero_stats_kernel(float* __restrict__ gsum, float* __restrict__ gsq) {
  int i = threadIdx.x;
  if (i < FEAT) { gsum[i] = 0.0f; gsq[i] = 0.0f; }
}

__global__ __launch_bounds__(128)
void bn_stats_kernel(const float* __restrict__ h, float* __restrict__ gsum,
                     float* __restrict__ gsq, int N) {
  int f = threadIdx.x;                 // feature
  long r0 = (long)blockIdx.x * 256;
  float s = 0.0f, q = 0.0f;
  for (int i = 0; i < 256; i++) {
    long r = r0 + i;
    if (r < N) {
      float v = h[r * FEAT + f];
      s += v;
      q += v * v;
    }
  }
  atomicAdd(&gsum[f], s);
  atomicAdd(&gsq[f], q);
}

__global__ __launch_bounds__(128)
void bn_finalize_kernel(const float* __restrict__ gsum, const float* __restrict__ gsq,
                        const float* __restrict__ gamma, const float* __restrict__ beta,
                        float* __restrict__ scale, float* __restrict__ shift, int N) {
  int f = threadIdx.x;
  float inv_n = 1.0f / (float)N;
  float mean = gsum[f] * inv_n;
  float var  = gsq[f] * inv_n - mean * mean;   // biased variance
  float sc   = rsqrtf(var + BN_EPS) * gamma[f];
  scale[f] = sc;
  shift[f] = beta[f] - mean * sc;
}

__global__ void bn_apply_elu_kernel(float* __restrict__ h, const float* __restrict__ scale,
                                    const float* __restrict__ shift, long total) {
  long idx = (long)blockIdx.x * blockDim.x + threadIdx.x;
  if (idx >= total) return;
  int f = (int)(idx & (FEAT - 1));
  float v = fmaf(h[idx], scale[f], shift[f]);
  h[idx] = v > 0.0f ? v : expm1f(v);  // ELU, alpha = 1
}

extern "C" void kernel_launch(void* const* d_in, const int* in_sizes, int n_in,
                              void* d_out, int out_size, void* d_ws, size_t ws_size,
                              hipStream_t stream) {
  const float* x   = (const float*)d_in[0];
  const int*  eidx = (const int*)d_in[1];   // [2,E] int32 (jax x64 disabled)
  const float* W1  = (const float*)d_in[2];
  const float* b1  = (const float*)d_in[3];
  const float* g1  = (const float*)d_in[4];
  const float* be1 = (const float*)d_in[5];
  const float* W2  = (const float*)d_in[6];
  const float* b2  = (const float*)d_in[7];
  const float* g2  = (const float*)d_in[8];
  const float* be2 = (const float*)d_in[9];
  float* out = (float*)d_out;

  const int N = in_sizes[0] / FEAT;
  const int E = in_sizes[1] / 2;
  const int* src = eidx;
  const int* dst = eidx + E;
  const long total = (long)N * FEAT;

  // workspace carve (~103 MB)
  float* buf_m   = (float*)d_ws;                   // N*128 : projection result / gather source
  float* buf_agg = buf_m + (size_t)N * FEAT;       // N*128 : layer-1 aggregate / h1
  float* dis     = buf_agg + (size_t)N * FEAT;     // N     : deg -> deg^-1/2
  float* gsum    = dis + N;                        // 128
  float* gsq     = gsum + FEAT;                    // 128
  float* scale   = gsq + FEAT;                     // 128
  float* shift   = scale + FEAT;                   // 128

  const int gN  = (N + 255) / 256;
  const int gE  = (E + 255) / 256;
  const int gG  = (N + 127) / 128;                 // GEMM blocks (128 rows each)
  const int gS  = (E + 1) / 2;                     // scatter blocks (2 edges each)
  const int gT  = (int)((total + 255) / 256);
  const int gB  = (N + 255) / 256;                 // bn-stats blocks (256 rows each)

  // ---- norm coefficients ----
  init_deg_kernel<<<gN, 256, 0, stream>>>(dis, N);
  accum_deg_kernel<<<gE, 256, 0, stream>>>(dst, dis, E);
  deg_to_dis_kernel<<<gN, 256, 0, stream>>>(dis, N);

  // ---- layer 1 ----
  gemm_wmma_kernel<<<gG, 256, 0, stream>>>(x, W1, buf_m, N);
  init_agg_kernel<<<gT, 256, 0, stream>>>(buf_m, dis, b1, buf_agg, total);
  scatter_edges_kernel<<<gS, 256, 0, stream>>>(buf_m, src, dst, dis, buf_agg, E);
  zero_stats_kernel<<<1, 256, 0, stream>>>(gsum, gsq);
  bn_stats_kernel<<<gB, 128, 0, stream>>>(buf_agg, gsum, gsq, N);
  bn_finalize_kernel<<<1, 128, 0, stream>>>(gsum, gsq, g1, be1, scale, shift, N);
  bn_apply_elu_kernel<<<gT, 256, 0, stream>>>(buf_agg, scale, shift, total);

  // ---- layer 2 (aggregate straight into d_out) ----
  gemm_wmma_kernel<<<gG, 256, 0, stream>>>(buf_agg, W2, buf_m, N);
  init_agg_kernel<<<gT, 256, 0, stream>>>(buf_m, dis, b2, out, total);
  scatter_edges_kernel<<<gS, 256, 0, stream>>>(buf_m, src, dst, dis, out, E);
  zero_stats_kernel<<<1, 256, 0, stream>>>(gsum, gsq);
  bn_stats_kernel<<<gB, 128, 0, stream>>>(out, gsum, gsq, N);
  bn_finalize_kernel<<<1, 128, 0, stream>>>(gsum, gsq, g2, be2, scale, shift, N);
  bn_apply_elu_kernel<<<gT, 256, 0, stream>>>(out, scale, shift, total);
}
